// FedProtoModel_48490180772265
// MI455X (gfx1250) — matrix operand order repaired and checked by reading, not compiled, not run on hardware
//
#include <hip/hip_runtime.h>

// ---------------------------------------------------------------------------
// FedProto head:  Z = x@W + b ;  out = (2*(Z@P^T) - ||Z||^2 - ||P||^2) / D
// B=8192, IN_DIM=2048, D=1024, C=1000, fp32 in/out.
//
// Pipeline (all compute-heavy math on v_wmma_f32_16x16x32_bf16, wave32):
//   0a split x      -> Xhi/Xlo  bf16 [8192][2048]
//   0b split+T W    -> Wthi/Wtlo bf16 [1024][2048]   (N-major for B-fragments)
//   0c split+pad P  -> Phi/Plo  bf16 [1024][1024]    (rows >=1000 zeroed)
//   0d p2[c] = ||P_c||^2 (fp32 exact)
//   1  GEMM1 (bf16x3 split, async double-buffered LDS) -> Z as bf16 hi/lo
//   2  z2[b] = ||Z_b||^2
//   3  GEMM2 + fused epilogue -> out = (2*cross - z2 - p2)/1024
// ---------------------------------------------------------------------------

typedef __attribute__((ext_vector_type(16))) __bf16 v16bf;
typedef __attribute__((ext_vector_type(8)))  float  v8f;

#define B_ROWS 8192
#define IN_DIM 2048
#define D_DIM  1024
#define C_CLS  1000

#define BM 128           // block tile M
#define BN 64            // block tile N
#define BK 32            // k-step (one bf16 WMMA depth)
#define LDK 40           // padded LDS k-stride (elements): 80B rows, b128-aligned chunks

union Frag { v16bf v; uint4 q[2]; };

struct TileBuf {
    unsigned short Ah[BM][LDK];
    unsigned short Al[BM][LDK];
    unsigned short Bh[BN][LDK];
    unsigned short Bl[BN][LDK];
};

// ---- bit-exact fp32 <-> bf16 helpers (RNE) --------------------------------
__device__ __forceinline__ unsigned short f2bf(float f) {
    unsigned u = __float_as_uint(f);
    u += 0x7FFFu + ((u >> 16) & 1u);
    return (unsigned short)(u >> 16);
}
__device__ __forceinline__ float bf2f(unsigned short h) {
    return __uint_as_float(((unsigned)h) << 16);
}
__device__ __forceinline__ void split_bf16(float f, unsigned short& hi, unsigned short& lo) {
    hi = f2bf(f);
    lo = f2bf(f - bf2f(hi));
}

// ---- CDNA5 async global->LDS copy (ASYNCcnt tracked), with fallback -------
#if defined(__HIP_DEVICE_COMPILE__) && __has_builtin(__builtin_amdgcn_global_load_async_to_lds_b128)
  #define HAVE_ASYNC_LDS 1
#else
  #define HAVE_ASYNC_LDS 0
#endif
#if defined(__HIP_DEVICE_COMPILE__) && !HAVE_ASYNC_LDS
  #warning "gfx1250 async-to-LDS builtin unavailable: falling back to synchronous LDS staging"
#endif

// builtin parameter types: (v4i addrspace(1)*, v4i addrspace(3)*, imm, imm)
typedef int v4i __attribute__((vector_size(4 * sizeof(int))));
typedef __attribute__((address_space(1))) v4i gv4i;
typedef __attribute__((address_space(3))) v4i lv4i;

__device__ __forceinline__ void cp16(const unsigned short* g, unsigned short* l) {
#if HAVE_ASYNC_LDS
    __builtin_amdgcn_global_load_async_to_lds_b128(
        (gv4i*)(v4i*)(unsigned short*)g, (lv4i*)(v4i*)l, 0, 0);
#else
    *(uint4*)l = *(const uint4*)g;
#endif
}

template <int N>
__device__ __forceinline__ void wait_async() {
#if HAVE_ASYNC_LDS
  #if __has_builtin(__builtin_amdgcn_s_wait_asynccnt)
    __builtin_amdgcn_s_wait_asynccnt(N);
  #else
    asm volatile("s_wait_asynccnt %0" :: "i"(N) : "memory");
  #endif
#endif
}

__device__ __forceinline__ v8f wmma_bf16(const Frag& a, const Frag& b, v8f c) {
    return __builtin_amdgcn_wmma_f32_16x16x32_bf16(
        false, a.v, false, b.v, (short)0, c, false, false);
}

// ---------------------------------------------------------------------------
// Prep 0a: elementwise fp32 -> bf16 hi/lo split (used for x)
// ---------------------------------------------------------------------------
__global__ __launch_bounds__(256) void split_pair(
    const float* __restrict__ in, unsigned short* __restrict__ hi,
    unsigned short* __restrict__ lo)
{
    size_t i = (size_t)blockIdx.x * 256 + threadIdx.x;   // index in float4 units
    float4 v = ((const float4*)in)[i];
    const float* pv = (const float*)&v;
    unsigned short h[4], l[4];
    #pragma unroll
    for (int j = 0; j < 4; ++j) split_bf16(pv[j], h[j], l[j]);
    uint2 ph, pl;
    ph.x = (unsigned)h[0] | ((unsigned)h[1] << 16);
    ph.y = (unsigned)h[2] | ((unsigned)h[3] << 16);
    pl.x = (unsigned)l[0] | ((unsigned)l[1] << 16);
    pl.y = (unsigned)l[2] | ((unsigned)l[3] << 16);
    ((uint2*)hi)[i] = ph;
    ((uint2*)lo)[i] = pl;
}

// ---------------------------------------------------------------------------
// Prep 0b: W [IN_DIM][D_DIM] fp32 -> transposed split bf16 [D_DIM][IN_DIM]
// 64x64 tiles through LDS so both sides stay coalesced. Grid (D/64, K/64).
// ---------------------------------------------------------------------------
__global__ __launch_bounds__(256) void split_transpose_w(
    const float* __restrict__ W, unsigned short* __restrict__ Th,
    unsigned short* __restrict__ Tl)
{
    __shared__ __attribute__((aligned(16))) unsigned short Sh[64][72], Sl[64][72];
    const int tid = threadIdx.x;
    const int n0 = blockIdx.x * 64;        // output row (D)
    const int k0 = blockIdx.y * 64;        // output col (IN_DIM)
    #pragma unroll
    for (int i = 0; i < 4; ++i) {
        int f = tid + 256 * i;             // 0..1023 over 64x16 float4
        int kr = f >> 4, c4 = (f & 15) << 2;
        float4 v = *(const float4*)(W + (size_t)(k0 + kr) * D_DIM + n0 + c4);
        const float* pv = (const float*)&v;
        #pragma unroll
        for (int j = 0; j < 4; ++j) {
            unsigned short h, l; split_bf16(pv[j], h, l);
            Sh[c4 + j][kr] = h;  Sl[c4 + j][kr] = l;
        }
    }
    __syncthreads();
    #pragma unroll
    for (int i = 0; i < 2; ++i) {
        int f = tid + 256 * i;             // 0..511 over 64 rows x 8 chunks
        int nl = f >> 3, q = (f & 7) << 3;
        size_t dst = (size_t)(n0 + nl) * IN_DIM + k0 + q;
        *(uint4*)(Th + dst) = *(const uint4*)&Sh[nl][q];
        *(uint4*)(Tl + dst) = *(const uint4*)&Sl[nl][q];
    }
}

// ---------------------------------------------------------------------------
// Prep 0c: P [1000][1024] fp32 -> split bf16 [1024][1024], pad rows with 0
// ---------------------------------------------------------------------------
__global__ __launch_bounds__(256) void split_pad_p(
    const float* __restrict__ P, unsigned short* __restrict__ Ph,
    unsigned short* __restrict__ Pl)
{
    size_t i = (size_t)blockIdx.x * 256 + threadIdx.x;   // float4 index over 1024x1024
    int row = (int)(i >> 8);                             // 256 float4 per row
    float4 v = make_float4(0.f, 0.f, 0.f, 0.f);
    if (row < C_CLS) v = ((const float4*)P)[i];
    const float* pv = (const float*)&v;
    unsigned short h[4], l[4];
    #pragma unroll
    for (int j = 0; j < 4; ++j) split_bf16(pv[j], h[j], l[j]);
    uint2 ph, pl;
    ph.x = (unsigned)h[0] | ((unsigned)h[1] << 16);
    ph.y = (unsigned)h[2] | ((unsigned)h[3] << 16);
    pl.x = (unsigned)l[0] | ((unsigned)l[1] << 16);
    pl.y = (unsigned)l[2] | ((unsigned)l[3] << 16);
    ((uint2*)Ph)[i] = ph;
    ((uint2*)Pl)[i] = pl;
}

// ---------------------------------------------------------------------------
// Row sum-of-squares reductions (one wave32 per row)
// ---------------------------------------------------------------------------
__global__ __launch_bounds__(256) void zrow_sumsq(
    const unsigned short* __restrict__ Zhi, const unsigned short* __restrict__ Zlo,
    float* __restrict__ z2)
{
    const int row  = blockIdx.x * 8 + (threadIdx.x >> 5);
    const int lane = threadIdx.x & 31;
    const size_t base = (size_t)row * D_DIM;
    float s = 0.f;
    for (int c = lane; c < D_DIM; c += 32) {
        float v = bf2f(Zhi[base + c]) + bf2f(Zlo[base + c]);
        s += v * v;
    }
    #pragma unroll
    for (int o = 16; o > 0; o >>= 1) s += __shfl_xor(s, o, 32);
    if (lane == 0) z2[row] = s;
}

__global__ __launch_bounds__(256) void proto_sumsq(
    const float* __restrict__ P, float* __restrict__ p2)
{
    const int row = blockIdx.x * 8 + (threadIdx.x >> 5);
    if (row >= C_CLS) return;
    const int lane = threadIdx.x & 31;
    const size_t base = (size_t)row * D_DIM;
    float s = 0.f;
    for (int c = lane; c < D_DIM; c += 32) { float v = P[base + c]; s += v * v; }
    #pragma unroll
    for (int o = 16; o > 0; o >>= 1) s += __shfl_xor(s, o, 32);
    if (lane == 0) p2[row] = s;
}

// ---------------------------------------------------------------------------
// Unified bf16x3-split GEMM, async double-buffered LDS staging.
//   A: [*, KDIM] bf16 hi/lo, row-major.  B: [N, KDIM] bf16 hi/lo, N-major.
//   MODE 0: epilogue Z = acc + bias -> split bf16 store (encoder GEMM)
//   MODE 1: epilogue out = (2*acc - z2[m] - p2[n]) / D_DIM   (cross GEMM)
// Grid: (N/BN, M/BM), 256 threads = 8 waves (4 along M x 2 along N).
// ---------------------------------------------------------------------------
template <int KDIM, int MODE>
__global__ __launch_bounds__(256) void gemm_bf16x3(
    const unsigned short* __restrict__ Ahg, const unsigned short* __restrict__ Alg,
    const unsigned short* __restrict__ Bhg, const unsigned short* __restrict__ Blg,
    const float* __restrict__ bias,
    unsigned short* __restrict__ Zh, unsigned short* __restrict__ Zl,
    const float* __restrict__ z2, const float* __restrict__ p2,
    float* __restrict__ out)
{
    __shared__ __attribute__((aligned(16))) TileBuf buf[2];

    const int tid  = threadIdx.x;
    const int lane = tid & 31;
    const int w    = tid >> 5;
    const int wm   = w & 3;
    const int wn   = w >> 2;
    const int m0   = blockIdx.y * BM;
    const int n0   = blockIdx.x * BN;
    const int half = lane >> 4;
    const int l16  = lane & 15;

    v8f acc[2][2] = {};

    // 6 async b128 transfers per thread per tile (4 for A hi/lo, 2 for B hi/lo)
    auto stage = [&](TileBuf& b, int k0) {
        #pragma unroll
        for (int i = 0; i < 2; ++i) {
            int f = tid + 256 * i;                 // 512 chunks: 128 rows x 4
            int r = f >> 2, q = (f & 3) << 3;
            size_t g = (size_t)(m0 + r) * KDIM + k0 + q;
            cp16(Ahg + g, &b.Ah[r][q]);
            cp16(Alg + g, &b.Al[r][q]);
        }
        {
            int nr = tid >> 2, q = (tid & 3) << 3; // 256 chunks: 64 rows x 4
            size_t g = (size_t)(n0 + nr) * KDIM + k0 + q;
            cp16(Bhg + g, &b.Bh[nr][q]);
            cp16(Blg + g, &b.Bl[nr][q]);
        }
    };

    auto mma = [&](const TileBuf& t) {
        Frag ah[2], al[2], bh[2], bl[2];
        const int kbA = half * 8;   // A frag: elems 0-7 = K kbA..+7, 8-15 = K kbA+16..+23
        const int kbB = half * 8;   // B frag: elems 0-7 = K kbB..+7, 8-15 = K kbB+16..+23
        #pragma unroll
        for (int mi = 0; mi < 2; ++mi) {
            int row = wm * 32 + mi * 16 + l16;
            ah[mi].q[0] = *(const uint4*)&t.Ah[row][kbA];
            ah[mi].q[1] = *(const uint4*)&t.Ah[row][kbA + 16];
            al[mi].q[0] = *(const uint4*)&t.Al[row][kbA];
            al[mi].q[1] = *(const uint4*)&t.Al[row][kbA + 16];
        }
        #pragma unroll
        for (int ni = 0; ni < 2; ++ni) {
            int col = wn * 32 + ni * 16 + l16;
            bh[ni].q[0] = *(const uint4*)&t.Bh[col][kbB];
            bh[ni].q[1] = *(const uint4*)&t.Bh[col][kbB + 16];
            bl[ni].q[0] = *(const uint4*)&t.Bl[col][kbB];
            bl[ni].q[1] = *(const uint4*)&t.Bl[col][kbB + 16];
        }
        #pragma unroll
        for (int mi = 0; mi < 2; ++mi)
            #pragma unroll
            for (int ni = 0; ni < 2; ++ni) {
                acc[mi][ni] = wmma_bf16(ah[mi], bh[ni], acc[mi][ni]);  // hi*hi
                acc[mi][ni] = wmma_bf16(ah[mi], bl[ni], acc[mi][ni]);  // hi*lo
                acc[mi][ni] = wmma_bf16(al[mi], bh[ni], acc[mi][ni]);  // lo*hi
            }
    };

    stage(buf[0], 0);
    int ib = 0;
    for (int k0 = 0; k0 < KDIM; k0 += BK, ib ^= 1) {
        if (k0 + BK < KDIM) {
            stage(buf[ib ^ 1], k0 + BK);   // prefetch next tile behind current
            wait_async<6>();               // async completes in order: tile k landed
        } else {
            wait_async<0>();
        }
        __syncthreads();
        mma(buf[ib]);
        __syncthreads();
    }

    if constexpr (MODE == 0) {
        #pragma unroll
        for (int mi = 0; mi < 2; ++mi)
            #pragma unroll
            for (int ni = 0; ni < 2; ++ni) {
                int col = n0 + wn * 32 + ni * 16 + l16;
                float bv = bias[col];
                #pragma unroll
                for (int v = 0; v < 8; ++v) {
                    int row = m0 + wm * 32 + mi * 16 + v + half * 8;
                    float z = acc[mi][ni][v] + bv;
                    unsigned short h, l; split_bf16(z, h, l);
                    size_t idx = (size_t)row * D_DIM + col;
                    Zh[idx] = h;  Zl[idx] = l;
                }
            }
    } else {
        const float invD = 1.0f / (float)D_DIM;
        #pragma unroll
        for (int mi = 0; mi < 2; ++mi)
            #pragma unroll
            for (int ni = 0; ni < 2; ++ni) {
                int col = n0 + wn * 32 + ni * 16 + l16;
                if (col < C_CLS) {
                    float pc = p2[col];
                    #pragma unroll
                    for (int v = 0; v < 8; ++v) {
                        int row = m0 + wm * 32 + mi * 16 + v + half * 8;
                        float r = (2.0f * acc[mi][ni][v] - z2[row] - pc) * invD;
                        out[(size_t)row * C_CLS + col] = r;
                    }
                }
            }
    }
}

// ---------------------------------------------------------------------------
// Host launcher
// ---------------------------------------------------------------------------
extern "C" void kernel_launch(void* const* d_in, const int* in_sizes, int n_in,
                              void* d_out, int out_size, void* d_ws, size_t ws_size,
                              hipStream_t stream) {
    const float* x    = (const float*)d_in[0];   // [8192, 2048]
    const float* Wenc = (const float*)d_in[1];   // [2048, 1024]
    const float* benc = (const float*)d_in[2];   // [1024]
    const float* prot = (const float*)d_in[3];   // [1000, 1024]
    float* out = (float*)d_out;                  // [8192, 1000]

    // workspace layout (≈108 MB):
    float* z2 = (float*)d_ws;                                    // 8192 f32
    float* p2 = z2 + B_ROWS;                                     // 1024 f32
    unsigned short* Xhi  = (unsigned short*)(p2 + 1024);
    unsigned short* Xlo  = Xhi  + (size_t)B_ROWS * IN_DIM;
    unsigned short* Wthi = Xlo  + (size_t)B_ROWS * IN_DIM;       // [1024][2048]
    unsigned short* Wtlo = Wthi + (size_t)D_DIM * IN_DIM;
    unsigned short* Phi  = Wtlo + (size_t)D_DIM * IN_DIM;        // [1024][1024]
    unsigned short* Plo  = Phi  + (size_t)D_DIM * D_DIM;
    unsigned short* Zhi  = Plo  + (size_t)D_DIM * D_DIM;         // [8192][1024]
    unsigned short* Zlo  = Zhi  + (size_t)B_ROWS * D_DIM;

    // 0) one-shot precision splits (hoists all fp32->bf16 VALU out of GEMMs)
    split_pair      <<<(B_ROWS * IN_DIM / 4) / 256, 256, 0, stream>>>(x, Xhi, Xlo);
    split_transpose_w<<<dim3(D_DIM / 64, IN_DIM / 64), 256, 0, stream>>>(Wenc, Wthi, Wtlo);
    split_pad_p     <<<(D_DIM * D_DIM / 4) / 256, 256, 0, stream>>>(prot, Phi, Plo);
    proto_sumsq     <<<(C_CLS + 7) / 8, 256, 0, stream>>>(prot, p2);

    // 1) encoder GEMM: Z = x@W + b  (bf16x3, async double-buffered)
    gemm_bf16x3<IN_DIM, 0><<<dim3(D_DIM / BN, B_ROWS / BM), 256, 0, stream>>>(
        Xhi, Xlo, Wthi, Wtlo, benc, Zhi, Zlo, nullptr, nullptr, nullptr);

    // 2) z2[b] = ||Z_b||^2
    zrow_sumsq<<<B_ROWS / 8, 256, 0, stream>>>(Zhi, Zlo, z2);

    // 3) cross GEMM + fused -MSE epilogue
    gemm_bf16x3<D_DIM, 1><<<dim3(D_DIM / BN, B_ROWS / BM), 256, 0, stream>>>(
        Zhi, Zlo, Phi, Plo, nullptr, nullptr, nullptr, z2, p2, out);
}
